// MultiHeadAttention_54649163874766
// MI455X (gfx1250) — matrix operand order repaired and checked by reading, not compiled, not run on hardware
//
#include <hip/hip_runtime.h>
#include <hip/hip_bf16.h>

// ---------- types ----------
typedef __attribute__((ext_vector_type(16))) __bf16       v16bf;
typedef __attribute__((ext_vector_type(8)))  float        v8f;
typedef __attribute__((ext_vector_type(4)))  unsigned int u32x4;
typedef __attribute__((ext_vector_type(4)))  float        f32x4;
typedef unsigned short u16;

union Frag16 {
  u16   u[16];
  u32x4 q[2];
  v16bf v;
};

__device__ __forceinline__ u16 f2bf(float f) {
  unsigned int x = __builtin_bit_cast(unsigned int, f);
  unsigned int r = x + 0x7FFFu + ((x >> 16) & 1u);   // round-to-nearest-even
  return (u16)(r >> 16);
}
__device__ __forceinline__ unsigned int pack2bf(float a, float b) {
  return (unsigned int)f2bf(a) | ((unsigned int)f2bf(b) << 16);
}

// ---------- fragment loaders (CDNA5 wave32 WMMA layouts) ----------
// A fragment (16x32 bf16): lane<16: rows 0-15, K = kb..kb+7 and kb+16..kb+23 (kb=0)
//                          lane>=16: same rows, kb=8.
__device__ __forceinline__ v16bf load_a_bf16(const u16* __restrict__ src, int ld, int lane) {
  int row = lane & 15;
  int kb  = (lane < 16) ? 0 : 8;
  const u16* p = src + (long)row * ld + kb;
  Frag16 f;
  f.q[0] = *(const u32x4*)(p);        // K = kb .. kb+7
  f.q[1] = *(const u32x4*)(p + 16);   // K = kb+16 .. kb+23
  return f.v;
}

// B fragment (32x16 bf16): lane&15 = column N; K = kb..kb+7 and kb+16..kb+23 (kb=0 lanes<16, kb=8 lanes>=16).
// element (K=k, N=n) = src[n*ld + k]
__device__ __forceinline__ v16bf load_b_bf16(const u16* __restrict__ src, long ld, int lane) {
  int n  = lane & 15;
  int kb = (lane < 16) ? 0 : 8;
  const u16* p = src + (long)n * ld + kb;
  Frag16 f;
  f.q[0] = *(const u32x4*)(p);
  f.q[1] = *(const u32x4*)(p + 16);
  return f.v;
}

#define WMMA_BF16(A, B, C) \
  __builtin_amdgcn_wmma_f32_16x16x32_bf16(false, (A), false, (B), (short)0, (C), false, false)

// ---------------------------------------------------------------------------
// Kernel 0: one-time fp32 -> bf16 conversion (bandwidth-bound, RNE rounding).
// 8 elements per thread, vectorized 16B loads/stores.
// ---------------------------------------------------------------------------
__global__ void cvt_f32_to_bf16_kernel(const float* __restrict__ src,
                                       u16* __restrict__ dst) {
  long i = ((long)blockIdx.x * blockDim.x + threadIdx.x) * 8;
  f32x4 a = *(const f32x4*)(src + i);
  f32x4 b = *(const f32x4*)(src + i + 4);
  u32x4 o;
  o.x = pack2bf(a.x, a.y);
  o.y = pack2bf(a.z, a.w);
  o.z = pack2bf(b.x, b.y);
  o.w = pack2bf(b.z, b.w);
  *(u32x4*)(dst + i) = o;
}

// ---------------------------------------------------------------------------
// Kernel 1: qkv = x @ W_qkv^T  (pure bf16 WMMA, fp32 accum) + fused RoPE.
// One wave per 32x64 tile (2 A-frags x 4 B-frags -> 8 WMMAs per K=32 step).
// 64 cols == one (t,h) slice so RoPE pairs d/d+32 live in acc[r][i]/acc[r][i+2].
// Q,K stored [bh][s][64]; V stored transposed [bh][d][S].
// ---------------------------------------------------------------------------
__global__ void mha_qkv_rope_kernel(const u16* __restrict__ xb,
                                    const u16* __restrict__ wqkvb,
                                    u16* __restrict__ qbuf,
                                    u16* __restrict__ kbuf,
                                    u16* __restrict__ vtbuf) {
  const int lane = threadIdx.x & 31;
  const int m0 = blockIdx.x * 32;     // row of x (0..4095)
  const int c0 = blockIdx.y * 64;     // col of qkv (0..3071)

  v8f acc[2][4] = {{v8f{}, v8f{}, v8f{}, v8f{}}, {v8f{}, v8f{}, v8f{}, v8f{}}};

  for (int k0 = 0; k0 < 1024; k0 += 32) {
    v16bf a0 = load_a_bf16(xb + (long)m0 * 1024 + k0,        1024, lane);
    v16bf a1 = load_a_bf16(xb + (long)(m0 + 16) * 1024 + k0, 1024, lane);
#pragma unroll
    for (int t = 0; t < 4; ++t) {
      v16bf b = load_b_bf16(wqkvb + (long)(c0 + 16 * t) * 1024 + k0, 1024, lane);
      acc[0][t] = WMMA_BF16(a0, b, acc[0][t]);
      acc[1][t] = WMMA_BF16(a1, b, acc[1][t]);
    }
  }

  const int t3    = c0 >> 10;            // 0=q 1=k 2=v
  const int h     = (c0 & 1023) >> 6;    // head
  const int n     = lane & 15;           // C-frag column
  const int rbase = (lane < 16) ? 0 : 8; // C-frag row base
  const int b_    = m0 >> 11;            // batch (tile never crosses batch)

  if (t3 < 2) {
    // RoPE: d in [0,32) pairs with d+32  ->  acc[r][i] pairs acc[r][i+2]
#pragma unroll
    for (int i = 0; i < 2; ++i) {
      float inv_freq = __powf(10000.0f, -((float)(16 * i + n)) / 32.0f);
#pragma unroll
      for (int r = 0; r < 2; ++r) {
#pragma unroll
        for (int v = 0; v < 8; ++v) {
          int s = (m0 + 16 * r + v + rbase) & 2047;
          float sn, cs;
          __sincosf((float)s * inv_freq, &sn, &cs);
          float t1 = acc[r][i][v], t2 = acc[r][i + 2][v];
          acc[r][i][v]     = t1 * cs - t2 * sn;
          acc[r][i + 2][v] = t1 * sn + t2 * cs;
        }
      }
    }
    u16* dst = (t3 == 0 ? qbuf : kbuf) + (long)(b_ * 16 + h) * 2048 * 64;
#pragma unroll
    for (int i = 0; i < 4; ++i) {
      int d = 16 * i + n;
#pragma unroll
      for (int r = 0; r < 2; ++r)
#pragma unroll
        for (int v = 0; v < 8; ++v) {
          int s = (m0 + 16 * r + v + rbase) & 2047;
          dst[(long)s * 64 + d] = f2bf(acc[r][i][v]);
        }
    }
  } else {
    // V transposed: [bh][d][S]
    u16* dst = vtbuf + (long)(b_ * 16 + h) * 64 * 2048;
#pragma unroll
    for (int i = 0; i < 4; ++i) {
      int d = 16 * i + n;
#pragma unroll
      for (int r = 0; r < 2; ++r)
#pragma unroll
        for (int v = 0; v < 8; ++v) {
          int s = (m0 + 16 * r + v + rbase) & 2047;
          dst[(long)d * 2048 + s] = f2bf(acc[r][i][v]);
        }
    }
  }
}

// ---------------------------------------------------------------------------
// Kernel 2: causal flash attention. One wave per (b,h,16-query tile).
// Key blocks of 32: 4 WMMAs QK^T, fp32 online softmax (shfl_xor over the
// 16-lane half that holds each row), P through LDS (C-layout -> A-layout),
// 4 WMMAs P*V. Output stored bf16 as [b][s][h*64+d] for the final GEMM.
// ---------------------------------------------------------------------------
__global__ void mha_flash_attn_kernel(const u16* __restrict__ qbuf,
                                      const u16* __restrict__ kbuf,
                                      const u16* __restrict__ vtbuf,
                                      u16* __restrict__ attn) {
  __shared__ u16 Plds[16 * 32];

  const int lane = threadIdx.x & 31;
  const int q0 = (blockIdx.x & 127) * 16;
  const int bh = blockIdx.x >> 7;          // 0..31
  const int b_ = bh >> 4, h = bh & 15;

  const u16* Q = qbuf  + (long)bh * 2048 * 64;
  const u16* K = kbuf  + (long)bh * 2048 * 64;
  const u16* V = vtbuf + (long)bh * 64 * 2048;

  const int n     = lane & 15;
  const int rbase = (lane < 16) ? 0 : 8;
  const float NEG_INF = -__builtin_inff();

  // Q tile A-fragments (dims 0-31 and 32-63), loaded once
  v16bf qa0 = load_a_bf16(Q + (long)q0 * 64 + 0,  64, lane);
  v16bf qa1 = load_a_bf16(Q + (long)q0 * 64 + 32, 64, lane);

  v8f o[4] = {v8f{}, v8f{}, v8f{}, v8f{}};
  float mrun[8], lrun[8];
#pragma unroll
  for (int v = 0; v < 8; ++v) { mrun[v] = NEG_INF; lrun[v] = 0.0f; }

  const float scale = 0.125f;  // 1/sqrt(64)

  for (int k0 = 0; k0 <= q0 + 15; k0 += 32) {
    // ---- scores for 32 keys: two 16x16 C fragments ----
    v8f sc[2] = {v8f{}, v8f{}};
#pragma unroll
    for (int hf = 0; hf < 2; ++hf) {
      v16bf b0 = load_b_bf16(K + (long)(k0 + 16 * hf) * 64 + 0,  64, lane);
      v16bf b1 = load_b_bf16(K + (long)(k0 + 16 * hf) * 64 + 32, 64, lane);
      sc[hf] = WMMA_BF16(qa0, b0, sc[hf]);
      sc[hf] = WMMA_BF16(qa1, b1, sc[hf]);
    }
    // ---- causal mask + scale ----
#pragma unroll
    for (int hf = 0; hf < 2; ++hf) {
      int key = k0 + 16 * hf + n;
#pragma unroll
      for (int v = 0; v < 8; ++v) {
        int sq = q0 + v + rbase;
        sc[hf][v] = (key <= sq) ? sc[hf][v] * scale : NEG_INF;
      }
    }
    // ---- online softmax per row ----
#pragma unroll
    for (int v = 0; v < 8; ++v) {
      float mx = fmaxf(sc[0][v], sc[1][v]);
#pragma unroll
      for (int off = 1; off < 16; off <<= 1) mx = fmaxf(mx, __shfl_xor(mx, off, 32));
      float mnew  = fmaxf(mrun[v], mx);           // always finite (key k0 <= sq)
      float alpha = __expf(mrun[v] - mnew);
      mrun[v] = mnew;
      float p0 = __expf(sc[0][v] - mnew);
      float p1 = __expf(sc[1][v] - mnew);
      sc[0][v] = p0; sc[1][v] = p1;
      float rs = p0 + p1;
#pragma unroll
      for (int off = 1; off < 16; off <<= 1) rs += __shfl_xor(rs, off, 32);
      lrun[v] = lrun[v] * alpha + rs;
#pragma unroll
      for (int i = 0; i < 4; ++i) o[i][v] *= alpha;
    }
    // ---- P: C-layout -> LDS (row-major 16x32 bf16) -> A-layout ----
#pragma unroll
    for (int hf = 0; hf < 2; ++hf)
#pragma unroll
      for (int v = 0; v < 8; ++v)
        Plds[(v + rbase) * 32 + 16 * hf + n] = f2bf(sc[hf][v]);
    __syncthreads();
    Frag16 pa;
    {
      int row = lane & 15;
      int kb  = (lane < 16) ? 0 : 8;
      const u16* p = Plds + row * 32 + kb;
      pa.q[0] = *(const u32x4*)(p);
      pa.q[1] = *(const u32x4*)(p + 16);
    }
    __syncthreads();
    // ---- O += P (16x32) @ V (32x64) : 4 WMMAs, V^T rows are contiguous ----
#pragma unroll
    for (int i = 0; i < 4; ++i) {
      v16bf vb = load_b_bf16(V + (long)(16 * i) * 2048 + k0, 2048, lane);
      o[i] = WMMA_BF16(pa.v, vb, o[i]);
    }
  }

  // ---- normalize + store bf16 to [b][s][h*64+d] ----
  u16* dst = attn + (long)b_ * 2048 * 1024 + (long)h * 64;
#pragma unroll
  for (int i = 0; i < 4; ++i) {
    int d = 16 * i + n;
#pragma unroll
    for (int v = 0; v < 8; ++v) {
      int s = q0 + v + rbase;
      dst[(long)s * 1024 + d] = f2bf(o[i][v] / lrun[v]);
    }
  }
}

// ---------------------------------------------------------------------------
// Kernel 3: out = attn @ W_out^T  (pure bf16 WMMA, fp32 out), 32x64 tiles.
// ---------------------------------------------------------------------------
__global__ void mha_out_proj_kernel(const u16* __restrict__ attn,
                                    const u16* __restrict__ woutb,
                                    float* __restrict__ out) {
  const int lane = threadIdx.x & 31;
  const int m0 = blockIdx.x * 32;
  const int c0 = blockIdx.y * 64;

  v8f acc[2][4] = {{v8f{}, v8f{}, v8f{}, v8f{}}, {v8f{}, v8f{}, v8f{}, v8f{}}};
  for (int k0 = 0; k0 < 1024; k0 += 32) {
    v16bf a0 = load_a_bf16(attn + (long)m0 * 1024 + k0,        1024, lane);
    v16bf a1 = load_a_bf16(attn + (long)(m0 + 16) * 1024 + k0, 1024, lane);
#pragma unroll
    for (int t = 0; t < 4; ++t) {
      v16bf b = load_b_bf16(woutb + (long)(c0 + 16 * t) * 1024 + k0, 1024, lane);
      acc[0][t] = WMMA_BF16(a0, b, acc[0][t]);
      acc[1][t] = WMMA_BF16(a1, b, acc[1][t]);
    }
  }
  const int n     = lane & 15;
  const int rbase = (lane < 16) ? 0 : 8;
#pragma unroll
  for (int i = 0; i < 4; ++i)
#pragma unroll
    for (int r = 0; r < 2; ++r)
#pragma unroll
      for (int v = 0; v < 8; ++v)
        out[(long)(m0 + 16 * r + v + rbase) * 1024 + c0 + 16 * i + n] = acc[r][i][v];
}

// ---------------------------------------------------------------------------
extern "C" void kernel_launch(void* const* d_in, const int* in_sizes, int n_in,
                              void* d_out, int out_size, void* d_ws, size_t ws_size,
                              hipStream_t stream) {
  const float* x    = (const float*)d_in[0];   // [2,2048,1024]
  const float* Wqkv = (const float*)d_in[1];   // [3072,1024]
  const float* Wout = (const float*)d_in[2];   // [1024,1024]
  float*       out  = (float*)d_out;           // [2,2048,1024]

  // workspace layout (u16 elements)
  const long XSZ = 4194304;   // 2*2048*1024
  const long WQ  = 3145728;   // 3072*1024
  const long WO  = 1048576;   // 1024*1024
  const long SEG = 4194304;   // per q/k/vt/attn buffer
  u16* xb    = (u16*)d_ws;
  u16* wqkvb = xb + XSZ;
  u16* woutb = wqkvb + WQ;
  u16* qbuf  = woutb + WO;
  u16* kbuf  = qbuf + SEG;
  u16* vtbuf = kbuf + SEG;
  u16* attn  = vtbuf + SEG;

  // 0) one-time fp32 -> bf16 conversion (2048 elems per 256-thread block)
  cvt_f32_to_bf16_kernel<<<dim3(XSZ / 2048), 256, 0, stream>>>(x, xb);
  cvt_f32_to_bf16_kernel<<<dim3(WQ / 2048), 256, 0, stream>>>(Wqkv, wqkvb);
  cvt_f32_to_bf16_kernel<<<dim3(WO / 2048), 256, 0, stream>>>(Wout, woutb);

  // 1) QKV projection + RoPE   (4096/32 row tiles x 3072/64 col tiles)
  mha_qkv_rope_kernel<<<dim3(128, 48), 32, 0, stream>>>(xb, wqkvb, qbuf, kbuf, vtbuf);

  // 2) flash attention         (2 batches * 16 heads * 128 q-tiles)
  mha_flash_attn_kernel<<<dim3(4096), 32, 0, stream>>>(qbuf, kbuf, vtbuf, attn);

  // 3) output projection       (4096/32 x 1024/64)
  mha_out_proj_kernel<<<dim3(128, 16), 32, 0, stream>>>(attn, woutb, out);
}
